// SparseMoE_506806141653
// MI455X (gfx1250) — matrix operand order repaired and checked by reading, not compiled, not run on hardware
//
#include <hip/hip_runtime.h>
#include <hip/hip_bf16.h>

#define N_TOKENS 2048
#define D_IN     1024
#define D_HID    4096
#define D_OUT    1024
#define N_EXPERTS 8
#define TM       64     // token tile per block (4 row-tiles of 16)
#define HCHUNK   1024   // hidden chunk per block (grid z dimension)
#define NTHREADS 512    // 16 waves = 2 row-pair groups x 8 col groups

typedef __attribute__((ext_vector_type(16))) __bf16 v16bf;
typedef __attribute__((ext_vector_type(8)))  __bf16 v8bf;
typedef __attribute__((ext_vector_type(8)))  float  v8f;

// ---------------------------------------------------------------------------
// Kernel 1: zero output accumulator and per-expert counters
// ---------------------------------------------------------------------------
__global__ void moe_init(float* __restrict__ out, int* __restrict__ counts) {
    const int i = blockIdx.x * blockDim.x + threadIdx.x;
    if (i < N_TOKENS * D_OUT) out[i] = 0.0f;
    if (i < N_EXPERTS) counts[i] = 0;
}

// ---------------------------------------------------------------------------
// Kernel 2: router — one wave32 per token: logits, softmax, top-2, scatter
// ---------------------------------------------------------------------------
__global__ __launch_bounds__(256) void moe_router(
    const float* __restrict__ x, const float* __restrict__ Wr,
    const float* __restrict__ br, int* __restrict__ counts,
    int* __restrict__ tok_idx, float* __restrict__ tok_w)
{
    const int lane = threadIdx.x & 31;
    const int wave = threadIdx.x >> 5;
    const int t = blockIdx.x * 8 + wave;
    if (t >= N_TOKENS) return;

    float acc[N_EXPERTS];
    #pragma unroll
    for (int e = 0; e < N_EXPERTS; ++e) acc[e] = 0.0f;

    const float* xr = x + (size_t)t * D_IN;
    for (int i = lane; i < D_IN; i += 32) {
        const float xv = xr[i];
        const float* wr = Wr + (size_t)i * N_EXPERTS;
        #pragma unroll
        for (int e = 0; e < N_EXPERTS; ++e) acc[e] += xv * wr[e];
    }
    // wave32 tree reduction
    #pragma unroll
    for (int e = 0; e < N_EXPERTS; ++e) {
        #pragma unroll
        for (int off = 16; off > 0; off >>= 1)
            acc[e] += __shfl_xor(acc[e], off, 32);
    }

    if (lane == 0) {
        float p[N_EXPERTS];
        float mx = -1e30f;
        #pragma unroll
        for (int e = 0; e < N_EXPERTS; ++e) {
            p[e] = acc[e] + br[e];
            mx = p[e] > mx ? p[e] : mx;
        }
        float s = 0.0f;
        #pragma unroll
        for (int e = 0; e < N_EXPERTS; ++e) { p[e] = __expf(p[e] - mx); s += p[e]; }
        const float inv = 1.0f / s;

        int e1 = 0;
        #pragma unroll
        for (int e = 1; e < N_EXPERTS; ++e) if (p[e] > p[e1]) e1 = e;   // first max on ties
        int e2 = (e1 == 0) ? 1 : 0;
        #pragma unroll
        for (int e = 0; e < N_EXPERTS; ++e) if (e != e1 && p[e] > p[e2]) e2 = e;

        int pos = atomicAdd(&counts[e1], 1);
        tok_idx[e1 * N_TOKENS + pos] = t;
        tok_w [e1 * N_TOKENS + pos] = p[e1] * inv;
        pos = atomicAdd(&counts[e2], 1);
        tok_idx[e2 * N_TOKENS + pos] = t;
        tok_w [e2 * N_TOKENS + pos] = p[e2] * inv;
    }
}

// ---------------------------------------------------------------------------
// WMMA fragment helpers (contiguous-K 16-bit operand order used by this code, wave32)
// A 16x32: lane = m + 16h -> row m, K = kc + 16h + {0..15}
// B 32x16: lane = n + 16h -> col n, K = kc + 16h + {0..15}
// ---------------------------------------------------------------------------
__device__ inline v16bf load_a_bf16(const __bf16* p /* &row[kc + 16h] */) {
    v8bf lo = *(const v8bf*)(p);
    v8bf hi = *(const v8bf*)(p + 8);
    return __builtin_shufflevector(lo, hi, 0,1,2,3,4,5,6,7,8,9,10,11,12,13,14,15);
}

__device__ inline v16bf load_b_f32(const float* p /* &W[k0+16h][col+n] */, int ld) {
    v16bf b;
    #pragma unroll
    for (int i = 0; i < 16; ++i) b[i] = (__bf16)p[(size_t)i * ld];
    return b;
}

// ---------------------------------------------------------------------------
// Kernel 3: fused expert FFN partial
//   grid: (32 token-tiles, 8 experts, 4 hidden-chunks), 512 threads.
//   Block computes, for its 64 gathered tokens and 1024-wide hidden slab:
//     Hc = relu(X @ W1[:, slab] + b1[slab])          (Phase A, via LDS)
//     Ypart = Hc @ W2[slab, :]                        (Phase B, regs)
//     out[token] += w * (Ypart + (slab==0 ? b2 : 0))  (atomic combine)
// ---------------------------------------------------------------------------
__global__ __launch_bounds__(NTHREADS) void moe_expert(
    const float* __restrict__ x,  const float* __restrict__ W1,
    const float* __restrict__ b1, const float* __restrict__ W2,
    const float* __restrict__ b2, const int* __restrict__ counts,
    const int* __restrict__ tok_idx, const float* __restrict__ tok_w,
    float* __restrict__ out)
{
    __shared__ __bf16 lds_x[TM * D_IN];    // 128 KB: gathered input tile
    __shared__ __bf16 lds_h[TM * HCHUNK];  // 128 KB: hidden-activation slab

    const int e     = blockIdx.y;
    const int hbase = blockIdx.z * HCHUNK;
    const int cnt   = counts[e];
    const int base  = blockIdx.x * TM;
    if (base >= cnt) return;

    const int tid  = threadIdx.x;
    const int lane = tid & 31;
    const int wave = tid >> 5;
    const int wR   = wave & 1;    // row-pair group: row-tiles {2wR, 2wR+1}
    const int wg   = wave >> 1;   // column-slice owner (0..7)

    // ---- warm up leading cachelines of this block's W1 slab (gfx1250
    //      global_prefetch_b8) while the X tile is staged to LDS ----
    {
        const float* w1s = W1 + (size_t)e * D_IN * D_HID + hbase;
        const int pr = tid >> 4;          // 32 rows
        const int pc = (tid & 15) << 6;   // 16 x 64-float segments per row
        __builtin_prefetch(w1s + (size_t)pr * D_HID + pc, 0, 1);
    }

    // ---- stage gathered X tile (fp32 -> bf16), zero-fill past count ----
    for (int idx = tid; idx < TM * D_IN; idx += NTHREADS) {
        const int row = idx >> 10;            // D_IN == 1024
        const int col = idx & (D_IN - 1);
        float v = 0.0f;
        if (base + row < cnt) {
            const int t = tok_idx[e * N_TOKENS + base + row];
            v = x[(size_t)t * D_IN + col];
        }
        lds_x[idx] = (__bf16)v;
    }
    __syncthreads();

    const int m  = lane & 15;   // A row / B col / C col selector
    const int n  = lane & 15;
    const int hh = lane >> 4;   // K-half (A/B) or M-half (C/D) selector

    v8f yacc[2][8];
    #pragma unroll
    for (int rr = 0; rr < 2; ++rr)
        #pragma unroll
        for (int j = 0; j < 8; ++j) yacc[rr][j] = {};

    // ---- Phase A: Hc = relu(X @ W1[:, hbase:+1024] + b1) -> LDS ----
    #pragma unroll 1
    for (int rr = 0; rr < 2; ++rr) {
        const int rt = wR * 2 + rr;
        const __bf16* xrow = lds_x + (rt * 16 + m) * D_IN + 16 * hh;
        #pragma unroll 1
        for (int ct = 0; ct < 8; ++ct) {
            const int ncol = (wg * 8 + ct) * 16;   // col within slab
            v8f acc = {};
            const float* w1p = W1 + (size_t)e * D_IN * D_HID
                             + (size_t)(16 * hh) * D_HID + (hbase + ncol + n);
            #pragma unroll 4
            for (int kk = 0; kk < D_IN; kk += 32) {
                v16bf a = load_a_bf16(xrow + kk);
                v16bf b = load_b_f32(w1p + (size_t)kk * D_HID, D_HID);
                acc = __builtin_amdgcn_wmma_f32_16x16x32_bf16(
                        false, a, false, b, (short)0, acc, false, false);
            }
            const float bias = b1[e * D_HID + hbase + ncol + n];
            #pragma unroll
            for (int r = 0; r < 8; ++r) {
                float v = acc[r] + bias;
                v = v > 0.0f ? v : 0.0f;             // ReLU
                lds_h[(rt * 16 + r + 8 * hh) * HCHUNK + ncol + n] = (__bf16)v;
            }
        }
    }
    __syncthreads();

    // ---- warm up leading cachelines of this block's W2 slab ----
    {
        const float* w2s = W2 + ((size_t)e * D_HID + hbase) * D_OUT;
        const int pr = tid >> 4;
        const int pc = (tid & 15) << 6;
        __builtin_prefetch(w2s + (size_t)pr * D_OUT + pc, 0, 1);
    }

    // ---- Phase B: Ypart += Hc @ W2[hbase:+1024, wave cols] ----
    #pragma unroll 1
    for (int rr = 0; rr < 2; ++rr) {
        const int rt = wR * 2 + rr;
        const __bf16* hrow = lds_h + (rt * 16 + m) * HCHUNK + 16 * hh;
        #pragma unroll 1
        for (int j = 0; j < 8; ++j) {
            const int ncol = wg * 128 + j * 16;
            const float* w2p = W2 + ((size_t)e * D_HID + hbase + 16 * hh) * D_OUT
                             + (ncol + n);
            v8f acc = yacc[rr][j];
            #pragma unroll 4
            for (int kk = 0; kk < HCHUNK; kk += 32) {
                v16bf a = load_a_bf16(hrow + kk);
                v16bf b = load_b_f32(w2p + (size_t)kk * D_OUT, D_OUT);
                acc = __builtin_amdgcn_wmma_f32_16x16x32_bf16(
                        false, a, false, b, (short)0, acc, false, false);
            }
            yacc[rr][j] = acc;
        }
    }

    // ---- epilogue: out[token] += w * (Ypart + b2?), guarded partial tile ----
    #pragma unroll 1
    for (int rr = 0; rr < 2; ++rr) {
        const int rt = wR * 2 + rr;
        #pragma unroll 1
        for (int j = 0; j < 8; ++j) {
            const int col = wg * 128 + j * 16 + n;
            const float bias = (hbase == 0) ? b2[e * D_OUT + col] : 0.0f;
            #pragma unroll
            for (int r = 0; r < 8; ++r) {
                const int row = rt * 16 + r + 8 * hh;  // lanes 16-31 hold M=8..15
                if (base + row < cnt) {
                    const int t = tok_idx[e * N_TOKENS + base + row];
                    const float w = tok_w[e * N_TOKENS + base + row];
                    atomicAdd(out + (size_t)t * D_OUT + col,
                              w * (yacc[rr][j][r] + bias));
                }
            }
        }
    }
}

// ---------------------------------------------------------------------------
extern "C" void kernel_launch(void* const* d_in, const int* in_sizes, int n_in,
                              void* d_out, int out_size, void* d_ws, size_t ws_size,
                              hipStream_t stream) {
    const float* x  = (const float*)d_in[0];
    const float* Wr = (const float*)d_in[1];
    const float* br = (const float*)d_in[2];
    const float* W1 = (const float*)d_in[3];
    const float* b1 = (const float*)d_in[4];
    const float* W2 = (const float*)d_in[5];
    const float* b2 = (const float*)d_in[6];
    float* out = (float*)d_out;

    char* ws = (char*)d_ws;
    int*   counts  = (int*)ws;                                      // 8 ints
    int*   tok_idx = (int*)(ws + 256);                              // 8*2048 ints
    float* tok_w   = (float*)(ws + 256 + sizeof(int) * N_EXPERTS * N_TOKENS);

    moe_init<<<(N_TOKENS * D_OUT + 255) / 256, 256, 0, stream>>>(out, counts);
    moe_router<<<N_TOKENS / 8, 256, 0, stream>>>(x, Wr, br, counts, tok_idx, tok_w);
    moe_expert<<<dim3(N_TOKENS / TM, N_EXPERTS, D_HID / HCHUNK), NTHREADS, 0, stream>>>(
        x, W1, b1, W2, b2, counts, tok_idx, tok_w, out);
}
